// Net_27968827031780
// MI455X (gfx1250) — matrix operand order repaired and checked by reading, compiled, not run on hardware
//
#include <hip/hip_runtime.h>
#include <hip/hip_bf16.h>
#include <math.h>

typedef __attribute__((ext_vector_type(16))) _Float16 v16h;
typedef __attribute__((ext_vector_type(8)))  float    v8f;

#define NBATCH 16384

// ---------------------------------------------------------------------------
// generic pad-convert: dst[Kp x Np] f16 = src[Kr x Nr] f32, zero elsewhere
// ---------------------------------------------------------------------------
__global__ void pad_cvt_kernel(const float* __restrict__ src,
                               _Float16* __restrict__ dst,
                               int Kr, int Nr, int Kp, int Np) {
    int i = blockIdx.x * blockDim.x + threadIdx.x;
    if (i >= Kp * Np) return;
    int k = i / Np, n = i % Np;
    dst[i] = (k < Kr && n < Nr) ? (_Float16)src[(size_t)k * Nr + n]
                                : (_Float16)0.f;
}

// plain fp32 -> f16 (same shape)
__global__ void f32_to_f16_kernel(const float* __restrict__ s,
                                  _Float16* __restrict__ d, int n) {
    int i = blockIdx.x * blockDim.x + threadIdx.x;
    if (i < n) d[i] = (_Float16)s[i];
}

// ---------------------------------------------------------------------------
// conv1 (3->6, 5x5 VALID) + relu + 2x2 maxpool, fp32 out [B,6,14,14]
// ---------------------------------------------------------------------------
__global__ void conv1_pool_kernel(const float* __restrict__ x,
                                  const float* __restrict__ w,
                                  const float* __restrict__ b,
                                  float* __restrict__ out, int total) {
    int o = blockIdx.x * blockDim.x + threadIdx.x;
    if (o >= total) return;
    int pw = o % 14, t1 = o / 14;
    int ph = t1 % 14, t2 = t1 / 14;
    int oc = t2 % 6,  n  = t2 / 6;
    const float* xb = x + (size_t)n * 3 * 32 * 32;
    const float* wc = w + oc * 75;
    float bias = b[oc];
    float mv = -3.4e38f;
    #pragma unroll
    for (int dy = 0; dy < 2; ++dy) {
        #pragma unroll
        for (int dx = 0; dx < 2; ++dx) {
            int oh = 2 * ph + dy, ow = 2 * pw + dx;
            float s = bias;
            #pragma unroll
            for (int ic = 0; ic < 3; ++ic) {
                const float* xc = xb + ic * 1024 + oh * 32 + ow;
                const float* wk = wc + ic * 25;
                #pragma unroll
                for (int kh = 0; kh < 5; ++kh)
                    #pragma unroll
                    for (int kw = 0; kw < 5; ++kw)
                        s = fmaf(xc[kh * 32 + kw], wk[kh * 5 + kw], s);
            }
            mv = fmaxf(mv, s);
        }
    }
    out[o] = fmaxf(mv, 0.f);
}

// ---------------------------------------------------------------------------
// conv2 (6->16, 5x5 VALID) + relu + 2x2 maxpool, f16 out flattened [B,416]
// (columns 400..415 zero-padded so fc1 can run K=416 branch-free)
// ---------------------------------------------------------------------------
__global__ void conv2_pool_kernel(const float* __restrict__ x1,
                                  const float* __restrict__ w,
                                  const float* __restrict__ b,
                                  _Float16* __restrict__ out, int total) {
    int o = blockIdx.x * blockDim.x + threadIdx.x;
    if (o >= total) return;
    int r = o % 416, n = o / 416;
    if (r >= 400) { out[o] = (_Float16)0.f; return; }
    int oc = r / 25, p = r % 25;
    int ph = p / 5,  pw = p % 5;
    const float* xb = x1 + (size_t)n * 6 * 14 * 14;
    const float* wc = w + oc * 150;
    float bias = b[oc];
    float mv = -3.4e38f;
    #pragma unroll
    for (int dy = 0; dy < 2; ++dy) {
        #pragma unroll
        for (int dx = 0; dx < 2; ++dx) {
            int oh = 2 * ph + dy, ow = 2 * pw + dx;
            float s = bias;
            #pragma unroll
            for (int ic = 0; ic < 6; ++ic) {
                const float* xc = xb + ic * 196 + oh * 14 + ow;
                const float* wk = wc + ic * 25;
                #pragma unroll
                for (int kh = 0; kh < 5; ++kh)
                    #pragma unroll
                    for (int kw = 0; kw < 5; ++kw)
                        s = fmaf(xc[kh * 14 + kw], wk[kh * 5 + kw], s);
            }
            mv = fmaxf(mv, s);
        }
    }
    out[o] = (_Float16)fmaxf(mv, 0.f);
}

// ---------------------------------------------------------------------------
// gate: comb = softmax(x @ gate_w)  (top-2 of 2 + renorm == identity)
// a2 is padded [B,128]; real width 84
// ---------------------------------------------------------------------------
__global__ void gate_kernel(const _Float16* __restrict__ x,
                            const float* __restrict__ gw,
                            float* __restrict__ comb, int nrows) {
    int n = blockIdx.x * blockDim.x + threadIdx.x;
    if (n >= nrows) return;
    const _Float16* xr = x + (size_t)n * 128;
    float l0 = 0.f, l1 = 0.f;
    #pragma unroll 4
    for (int i = 0; i < 84; ++i) {
        float xv = (float)xr[i];
        l0 = fmaf(xv, gw[i * 2 + 0], l0);
        l1 = fmaf(xv, gw[i * 2 + 1], l1);
    }
    float mx = fmaxf(l0, l1);
    float e0 = __expf(l0 - mx), e1 = __expf(l1 - mx);
    float inv = 1.f / (e0 + e1);
    comb[n * 2 + 0] = e0 * inv;
    comb[n * 2 + 1] = e1 * inv;
}

// ---------------------------------------------------------------------------
// WMMA GEMM: C[M,Np] = A[M,K](f16, packed lda==K) * B[K,Np](f16) + bias.
// Requirements (guaranteed by padded buffers): K % 32 == 0, Np % 64 == 0.
// Block: 256 threads (8 wave32 waves), 128x64 tile; wave w owns rows w*16..+16.
// MODE 0: f16 out, relu(acc+bias)      (padded cols self-zero)
// MODE 1: f32 out, (acc+bias)*scale[m]
// MODE 2: f32 out, += (acc+bias)*scale[m]
// MODE 3: f32 out, acc+bias
// ---------------------------------------------------------------------------
template <int MODE>
__global__ __launch_bounds__(256)
void gemm_wmma_kernel(const _Float16* __restrict__ A,
                      const _Float16* __restrict__ Bw, int ldb,
                      const float* __restrict__ bias,
                      int K, int Nbias, int Nstore,
                      void* __restrict__ Cout, int ldc,
                      const float* __restrict__ scale, int sstride) {
    // padded stride 40 halfs (80B) keeps 16B alignment for b128 LDS ops
    __shared__ _Float16 lds_a[128 * 40];  // [row][k]
    __shared__ _Float16 lds_bT[64 * 40];  // [n][k]

    const int t = threadIdx.x;
    const int m_block = blockIdx.y * 128;
    const int n_block = blockIdx.x * 64;
    const int lane = t & 31, wv = t >> 5;
    const int hi = lane >> 4, lm = lane & 15;

    union V8 { v8f v; float f[8]; };
    V8 acc[4];
    v8f zv = {0.f, 0.f, 0.f, 0.f, 0.f, 0.f, 0.f, 0.f};
    #pragma unroll
    for (int nt = 0; nt < 4; ++nt) acc[nt].v = zv;

    // A loader: 2 threads per row, 16 halfs (32B) each
    const int arl   = t >> 1;          // 0..127
    const int akseg = (t & 1) * 16;    // 0 or 16
    // B loader: 32 k x 8 n-segments of 8 halfs (16B)
    const int bkk   = t & 31;
    const int bnseg = (t >> 5) * 8;    // 0..56

    const _Float16* gA = A + (size_t)(m_block + arl) * K + akseg;
    const _Float16* gB = Bw + (size_t)bkk * ldb + n_block + bnseg;

    for (int k0 = 0; k0 < K; k0 += 32) {
        // ---- stage A[128x32] (aligned: K%16==0, akseg%16==0) ----
        {
            const uint4* g4 = (const uint4*)(gA + k0);
            uint4* l4 = (uint4*)&lds_a[arl * 40 + akseg];
            l4[0] = g4[0];
            l4[1] = g4[1];
        }
        // ---- stage B[32x64] transposed (aligned: ldb%64==0, nseg%8==0) ----
        {
            const _Float16* gb = gB + (size_t)k0 * ldb;
            if (k0 + 32 < K)
                __builtin_prefetch(gb + (size_t)32 * ldb, 0, 0);  // global_prefetch_b8
            uint4 rbuf = *(const uint4*)gb;
            const _Float16* tp = (const _Float16*)&rbuf;
            #pragma unroll
            for (int i = 0; i < 8; ++i)
                lds_bT[(bnseg + i) * 40 + bkk] = tp[i];
        }
        __syncthreads();

        // ---- per-lane fragments contiguous k order (k = 16*hi + s), same in A and B ----
        union F16x16 { v16h v; _Float16 h[16]; };
        F16x16 af;
        const _Float16* ap = &lds_a[(wv * 16 + lm) * 40];
        #pragma unroll
        for (int e = 0; e < 8; ++e) {
            af.h[e]     = ap[hi * 16 + e];       // K = e + 16*hi
            af.h[8 + e] = ap[hi * 16 + 8 + e];   // K = 8 + e + 16*hi
        }
        #pragma unroll
        for (int nt = 0; nt < 4; ++nt) {
            F16x16 bf;
            const _Float16* bp = &lds_bT[(nt * 16 + lm) * 40 + hi * 16];
            #pragma unroll
            for (int e = 0; e < 16; ++e) bf.h[e] = bp[e];    // K = e + 16*hi
            acc[nt].v = __builtin_amdgcn_wmma_f32_16x16x32_f16(
                false, af.v, false, bf.v, (short)0, acc[nt].v, false, false);
        }
        __syncthreads();
    }

    // ---- epilogue: lane holds (m = wave_base + hi*8 + r, n = lm) ----
    #pragma unroll
    for (int nt = 0; nt < 4; ++nt) {
        int n = n_block + nt * 16 + lm;
        if (n >= Nstore) continue;
        float bs = (n < Nbias) ? bias[n] : 0.f;
        #pragma unroll
        for (int r = 0; r < 8; ++r) {
            int m = m_block + wv * 16 + hi * 8 + r;
            float v = acc[nt].f[r] + bs;
            size_t idx = (size_t)m * ldc + n;
            if (MODE == 0) {
                ((_Float16*)Cout)[idx] = (_Float16)fmaxf(v, 0.f);
            } else if (MODE == 1) {
                ((float*)Cout)[idx] = v * scale[(size_t)m * sstride];
            } else if (MODE == 2) {
                ((float*)Cout)[idx] += v * scale[(size_t)m * sstride];
            } else {
                ((float*)Cout)[idx] = v;
            }
        }
    }
}

// ---------------------------------------------------------------------------
// host launcher
// ---------------------------------------------------------------------------
extern "C" void kernel_launch(void* const* d_in, const int* in_sizes, int n_in,
                              void* d_out, int out_size, void* d_ws, size_t ws_size,
                              hipStream_t stream) {
    const float* x       = (const float*)d_in[0];
    const float* conv1_w = (const float*)d_in[1];
    const float* conv1_b = (const float*)d_in[2];
    const float* conv2_w = (const float*)d_in[3];
    const float* conv2_b = (const float*)d_in[4];
    const float* fc1_w   = (const float*)d_in[5];
    const float* fc1_b   = (const float*)d_in[6];
    const float* fc2_w   = (const float*)d_in[7];
    const float* fc2_b   = (const float*)d_in[8];
    const float* gate_w  = (const float*)d_in[9];
    const float* exp_w1  = (const float*)d_in[10];
    const float* exp_b1  = (const float*)d_in[11];
    const float* exp_w2  = (const float*)d_in[12];
    const float* exp_b2  = (const float*)d_in[13];
    const float* fc3_w   = (const float*)d_in[14];
    const float* fc3_b   = (const float*)d_in[15];

    char* ws = (char*)d_ws;
    size_t off = 0;
    auto take = [&](size_t bytes) {
        size_t r = off;
        off += (bytes + 255) & ~(size_t)255;
        return r;
    };
    const size_t NB = NBATCH;
    size_t x1_off   = take(NB * 1176 * 4);   // conv1 out f32; aliased by h later
    size_t x2_off   = take(NB * 416 * 2);    // conv2 out f16 [B,416] (padded)
    size_t a1_off   = take(NB * 128 * 2);    // fc1 out f16 [B,128]
    size_t a2_off   = take(NB * 128 * 2);    // fc2 out f16 [B,128] (MoE input)
    size_t comb_off = take(NB * 2 * 4);      // gate probs f32
    size_t ymoe_off = take(NB * 128 * 4);    // MoE combined f32 [B,128]
    size_t ym16_off = take(NB * 128 * 2);    // MoE combined f16 [B,128]
    size_t wfc1_off = take((size_t)416 * 128 * 2);
    size_t wfc2_off = take((size_t)128 * 128 * 2);
    size_t we1_off  = take((size_t)2 * 128 * 2048 * 2);
    size_t we2_off  = take((size_t)2 * 2048 * 128 * 2);
    size_t wfc3_off = take((size_t)128 * 64 * 2);

    float*    x1   = (float*)(ws + x1_off);
    _Float16* x2   = (_Float16*)(ws + x2_off);
    _Float16* a1   = (_Float16*)(ws + a1_off);
    _Float16* a2   = (_Float16*)(ws + a2_off);
    float*    comb = (float*)(ws + comb_off);
    float*    ymoe = (float*)(ws + ymoe_off);
    _Float16* ym16 = (_Float16*)(ws + ym16_off);
    _Float16* h    = (_Float16*)(ws + x1_off);  // alias: conv1 act dead after conv2
    _Float16* wfc1 = (_Float16*)(ws + wfc1_off);
    _Float16* wfc2 = (_Float16*)(ws + wfc2_off);
    _Float16* we1  = (_Float16*)(ws + we1_off);
    _Float16* we2  = (_Float16*)(ws + we2_off);
    _Float16* wfc3 = (_Float16*)(ws + wfc3_off);

    auto padcvt = [&](const float* s, _Float16* d, int Kr, int Nr, int Kp, int Np) {
        int tot = Kp * Np;
        pad_cvt_kernel<<<(tot + 255) / 256, 256, 0, stream>>>(s, d, Kr, Nr, Kp, Np);
    };

    // weight pad-conversions (f32 -> zero-padded f16)
    padcvt(fc1_w, wfc1, 400, 120, 416, 128);
    padcvt(fc2_w, wfc2, 120, 84, 128, 128);
    padcvt(exp_w1,                      we1,                      84, 2048, 128, 2048);
    padcvt(exp_w1 + (size_t)84 * 2048,  we1 + (size_t)128 * 2048, 84, 2048, 128, 2048);
    padcvt(exp_w2,                      we2,                      2048, 84, 2048, 128);
    padcvt(exp_w2 + (size_t)2048 * 84,  we2 + (size_t)2048 * 128, 2048, 84, 2048, 128);
    padcvt(fc3_w, wfc3, 84, 10, 128, 64);

    // conv stages
    {
        int total = NB * 6 * 14 * 14;
        conv1_pool_kernel<<<(total + 255) / 256, 256, 0, stream>>>(
            x, conv1_w, conv1_b, x1, total);
    }
    {
        int total = NB * 416;
        conv2_pool_kernel<<<(total + 255) / 256, 256, 0, stream>>>(
            x1, conv2_w, conv2_b, x2, total);
    }

    dim3 blk(256);
    dim3 g2(128 / 64, NB / 128);     // Np = 128
    dim3 gH(2048 / 64, NB / 128);    // Np = 2048
    dim3 g1(64 / 64, NB / 128);      // Np = 64

    // fc1: [B,416] x [416,128] + relu -> a1
    gemm_wmma_kernel<0><<<g2, blk, 0, stream>>>(x2, wfc1, 128, fc1_b,
        416, 120, 128, a1, 128, nullptr, 0);
    // fc2: [B,128] x [128,128] + relu -> a2
    gemm_wmma_kernel<0><<<g2, blk, 0, stream>>>(a1, wfc2, 128, fc2_b,
        128, 84, 128, a2, 128, nullptr, 0);

    // gate softmax -> comb [B,2]
    gate_kernel<<<((int)NB + 255) / 256, 256, 0, stream>>>(a2, gate_w, comb, (int)NB);

    // expert 0
    gemm_wmma_kernel<0><<<gH, blk, 0, stream>>>(a2, we1, 2048, exp_b1,
        128, 2048, 2048, h, 2048, nullptr, 0);
    gemm_wmma_kernel<1><<<g2, blk, 0, stream>>>(h, we2, 128, exp_b2,
        2048, 84, 128, ymoe, 128, comb + 0, 2);
    // expert 1 (accumulate)
    gemm_wmma_kernel<0><<<gH, blk, 0, stream>>>(a2, we1 + (size_t)128 * 2048, 2048,
        exp_b1 + 2048, 128, 2048, 2048, h, 2048, nullptr, 0);
    gemm_wmma_kernel<2><<<g2, blk, 0, stream>>>(h, we2 + (size_t)2048 * 128, 128,
        exp_b2 + 84, 2048, 84, 128, ymoe, 128, comb + 1, 2);

    // fc3: [B,128] x [128,64] -> d_out f32 [B,10]
    f32_to_f16_kernel<<<((int)(NB * 128) + 255) / 256, 256, 0, stream>>>(
        ymoe, ym16, (int)(NB * 128));
    gemm_wmma_kernel<3><<<g1, blk, 0, stream>>>(ym16, wfc3, 64, fc3_b,
        128, 10, 10, (float*)d_out, 10, nullptr, 0);
}
